// TemporalGNN_37160057045540
// MI455X (gfx1250) — compile-verified, matrix operand order checked by reading, not run on hardware
//
#include <hip/hip_runtime.h>
#include <hip/hip_bf16.h>
#include <math.h>

// Sizes from the reference
#define TGN_T 12
#define TGN_N 20000
#define TGN_E 500000
#define TGN_F 128
#define TGN_H 128
#define TGN_L 2   // number of GCN convs (L-1)

typedef __bf16 bf16_t;
typedef __attribute__((ext_vector_type(16))) __bf16 v16bf;
typedef __attribute__((ext_vector_type(8)))  float  v8f;

// ---------------------------------------------------------------------------
// WMMA fragment loaders (wave32, 16-bit A 16x32 / B 32x16 layouts per ISA 7.12.2)
// A: lane L holds row M = L&15; elements 0..7 -> K = kb*32 + (L>>4)*8 + e,
//    elements 8..15 -> K = that + 16.  B (from pre-transposed weights Bt[col][K])
// uses the identical per-lane K pattern with "col" in place of "row".
// ---------------------------------------------------------------------------
__device__ __forceinline__ v16bf tgn_load_a_f32(const float* A, int lda, int row_base, int kb) {
  int lane = threadIdx.x & 31;
  int row  = row_base + (lane & 15);
  int k0   = kb * 32 + ((lane >> 4) << 3);
  const float* p0 = A + (size_t)row * lda + k0;
  const float* p1 = p0 + 16;
  v16bf a;
#pragma unroll
  for (int e = 0; e < 8; ++e) a[e] = (__bf16)p0[e];
#pragma unroll
  for (int e = 0; e < 8; ++e) a[8 + e] = (__bf16)p1[e];
  return a;
}

__device__ __forceinline__ v16bf tgn_load_b_bt(const bf16_t* Bt, int ldb, int col_base, int kb) {
  int lane = threadIdx.x & 31;
  int col  = col_base + (lane & 15);
  int k0   = kb * 32 + ((lane >> 4) << 3);
  const bf16_t* p0 = Bt + (size_t)col * ldb + k0;
  const bf16_t* p1 = p0 + 16;
  v16bf b;
#pragma unroll
  for (int e = 0; e < 8; ++e) b[e] = p0[e];
#pragma unroll
  for (int e = 0; e < 8; ++e) b[8 + e] = p1[e];
  return b;
}

__device__ __forceinline__ v8f tgn_wmma(v16bf a, v16bf b, v8f c) {
  return __builtin_amdgcn_wmma_f32_16x16x32_bf16(false, a, false, b, (short)0, c, false, false);
}

// ---------------------------------------------------------------------------
// Generic GEMM: C[M x ldc] = A[M x K] (f32, cvt->bf16) * Bt[Cout x K]^T + bias
// One wave (32 threads) per 16x16 output tile; K accumulated in steps of 32.
// ---------------------------------------------------------------------------
__global__ __launch_bounds__(32)
void tgn_gemm(const float* __restrict__ A, const bf16_t* __restrict__ Bt,
              const float* __restrict__ bias, float* __restrict__ C,
              int K, int ldc) {
  int mb = blockIdx.x * 16;
  int nb = blockIdx.y * 16;
  v8f acc = {};
  int nkb = K >> 5;
  for (int kb = 0; kb < nkb; ++kb) {
    v16bf a = tgn_load_a_f32(A, K, mb, kb);
    v16bf b = tgn_load_b_bt(Bt, K, nb, kb);
    acc = tgn_wmma(a, b, acc);
  }
  int lane = threadIdx.x & 31;
  int col  = nb + (lane & 15);
  float bv = bias ? bias[col] : 0.0f;
  int rb   = mb + ((lane >> 4) << 3);
#pragma unroll
  for (int r = 0; r < 8; ++r)
    C[(size_t)(rb + r) * ldc + col] = acc[r] + bv;
}

// ---------------------------------------------------------------------------
// Fused GRU cell: gi = Xp@W_ih + b_ih ; gh = H@W_hh + b_hh ; torch gate order
// (r,z,n). One wave computes a 16x16 tile of the new hidden state: 6 WMMA
// accumulator tiles (ir,iz,in,hr,hz,hn) then gate math in C-layout registers.
// ---------------------------------------------------------------------------
__global__ __launch_bounds__(32)
void tgn_gru(const float* __restrict__ Xp, const float* __restrict__ Hin,
             const bf16_t* __restrict__ Wt_ih, const bf16_t* __restrict__ Wt_hh,
             const float* __restrict__ b_ih, const float* __restrict__ b_hh,
             float* __restrict__ Hout) {
  int mb = blockIdx.x * 16;
  int nb = blockIdx.y * 16;
  v8f ir = {}, iz = {}, in_ = {}, hr = {}, hz = {}, hn = {};
  const int K = TGN_H;
#pragma unroll
  for (int kb = 0; kb < (K >> 5); ++kb) {
    v16bf ax = tgn_load_a_f32(Xp,  K, mb, kb);
    v16bf ah = tgn_load_a_f32(Hin, K, mb, kb);
    v16bf b;
    b = tgn_load_b_bt(Wt_ih,               K, nb, kb); ir  = tgn_wmma(ax, b, ir);
    b = tgn_load_b_bt(Wt_ih + 128 * TGN_H, K, nb, kb); iz  = tgn_wmma(ax, b, iz);
    b = tgn_load_b_bt(Wt_ih + 256 * TGN_H, K, nb, kb); in_ = tgn_wmma(ax, b, in_);
    b = tgn_load_b_bt(Wt_hh,               K, nb, kb); hr  = tgn_wmma(ah, b, hr);
    b = tgn_load_b_bt(Wt_hh + 128 * TGN_H, K, nb, kb); hz  = tgn_wmma(ah, b, hz);
    b = tgn_load_b_bt(Wt_hh + 256 * TGN_H, K, nb, kb); hn  = tgn_wmma(ah, b, hn);
  }
  int lane = threadIdx.x & 31;
  int col  = nb + (lane & 15);
  float bir = b_ih[col], biz = b_ih[col + 128], bin = b_ih[col + 256];
  float bhr = b_hh[col], bhz = b_hh[col + 128], bhn = b_hh[col + 256];
  int rb = mb + ((lane >> 4) << 3);
#pragma unroll
  for (int r = 0; r < 8; ++r) {
    int row = rb + r;
    float hprev = Hin[(size_t)row * TGN_H + col];
    float rg = 1.0f / (1.0f + expf(-(ir[r] + bir + hr[r] + bhr)));
    float zg = 1.0f / (1.0f + expf(-(iz[r] + biz + hz[r] + bhz)));
    float ng = tanhf(in_[r] + bin + rg * (hn[r] + bhn));
    Hout[(size_t)row * TGN_H + col] = (1.0f - zg) * ng + zg * hprev;
  }
}

// ---------------------------------------------------------------------------
// Weight prep: transpose f32 [K x Cout] -> bf16 [Cout x K] (contiguous-K rows
// so WMMA B fragments load as two 16B chunks per lane).
// ---------------------------------------------------------------------------
__global__ void tgn_prep_bt(const float* __restrict__ W, bf16_t* __restrict__ Wt,
                            int K, int Cout) {
  int idx = blockIdx.x * blockDim.x + threadIdx.x;
  if (idx < K * Cout) {
    int c = idx / K, k = idx - c * K;
    Wt[idx] = (__bf16)W[(size_t)k * Cout + c];
  }
}

__global__ void tgn_zero(float* __restrict__ p, int n) {
  int i = blockIdx.x * blockDim.x + threadIdx.x;
  if (i < n) p[i] = 0.0f;
}

// deg[dst] += ew  (self-loop +1 folded into dinv kernel)
__global__ void tgn_degree(const int* __restrict__ dst, const float* __restrict__ ew,
                           float* __restrict__ deg, int E) {
  int e = blockIdx.x * blockDim.x + threadIdx.x;
  if (e < E) atomicAdd(&deg[dst[e]], ew[e]);
}

__global__ void tgn_dinv(const float* __restrict__ deg, float* __restrict__ dinv, int n) {
  int i = blockIdx.x * blockDim.x + threadIdx.x;
  if (i < n) dinv[i] = rsqrtf(deg[i] + 1.0f);
}

// AGG[dst][h] += HW[src][h] * dinv[src]*ew*dinv[dst] ; 128 lanes per edge.
__global__ void tgn_scatter(const int* __restrict__ src, const int* __restrict__ dst,
                            const float* __restrict__ ew, const float* __restrict__ dinv,
                            const float* __restrict__ HW, float* __restrict__ AGG, int E) {
  long long t = (long long)blockIdx.x * blockDim.x + threadIdx.x;
  int e = (int)(t >> 7);
  int h = (int)(t & 127);
  if (e < E) {
    int s = src[e], d = dst[e];
    float coef = dinv[s] * ew[e] * dinv[d];
    atomicAdd(&AGG[(size_t)d * TGN_H + h], HW[(size_t)s * TGN_H + h] * coef);
  }
}

// Hout = relu(AGG + dinv^2*HW + conv_b) + Hin
__global__ void tgn_combine(const float* __restrict__ AGG, const float* __restrict__ HW,
                            const float* __restrict__ dinv, const float* __restrict__ cb,
                            const float* __restrict__ Hin, float* __restrict__ Hout, int n) {
  int t = blockIdx.x * blockDim.x + threadIdx.x;
  if (t < n) {
    int i = t >> 7, h = t & 127;
    float di = dinv[i];
    float v = AGG[t] + di * di * HW[t] + cb[h];
    Hout[t] = fmaxf(v, 0.0f) + Hin[t];
  }
}

// out[i] = dot(H[i,:], W_out) + b_out ; one wave per node.
__global__ void tgn_out(const float* __restrict__ H, const float* __restrict__ W_out,
                        const float* __restrict__ b_out, float* __restrict__ out, int N) {
  int node = (int)((blockIdx.x * (long long)blockDim.x + threadIdx.x) >> 5);
  int lane = threadIdx.x & 31;
  if (node < N) {
    const float* hr = H + (size_t)node * TGN_H;
    float s = 0.0f;
#pragma unroll
    for (int j = 0; j < TGN_H; j += 32) s += hr[j + lane] * W_out[j + lane];
#pragma unroll
    for (int off = 16; off > 0; off >>= 1) s += __shfl_down(s, off, 32);
    if (lane == 0) out[node] = s + b_out[0];
  }
}

// ---------------------------------------------------------------------------
extern "C" void kernel_launch(void* const* d_in, const int* in_sizes, int n_in,
                              void* d_out, int out_size, void* d_ws, size_t ws_size,
                              hipStream_t stream) {
  const float* x_seq       = (const float*)d_in[0];
  const int*   edge_index  = (const int*)  d_in[1];
  const float* edge_weight = (const float*)d_in[2];
  const float* W_in        = (const float*)d_in[3];
  const float* b_in        = (const float*)d_in[4];
  const float* W_ih        = (const float*)d_in[5];
  const float* b_ih        = (const float*)d_in[6];
  const float* W_hh        = (const float*)d_in[7];
  const float* b_hh        = (const float*)d_in[8];
  const float* conv_w      = (const float*)d_in[9];
  const float* conv_b      = (const float*)d_in[10];
  const float* W_out       = (const float*)d_in[11];
  const float* b_out       = (const float*)d_in[12];
  float* out = (float*)d_out;

  char* ws = (char*)d_ws;
  size_t off = 0;
  auto alloc = [&](size_t bytes) -> void* {
    void* p = ws + off;
    off += (bytes + 255) & ~(size_t)255;
    return p;
  };
  bf16_t* Wt_in   = (bf16_t*)alloc((size_t)TGN_F * TGN_H * 2);
  bf16_t* Wt_ih   = (bf16_t*)alloc((size_t)3 * TGN_H * TGN_H * 2);
  bf16_t* Wt_hh   = (bf16_t*)alloc((size_t)3 * TGN_H * TGN_H * 2);
  bf16_t* Wt_conv = (bf16_t*)alloc((size_t)TGN_L * TGN_H * TGN_H * 2);
  float* Ha   = (float*)alloc((size_t)TGN_N * TGN_H * 4);
  float* Hb   = (float*)alloc((size_t)TGN_N * TGN_H * 4);
  float* Xp   = (float*)alloc((size_t)TGN_N * TGN_H * 4);
  float* HW   = (float*)alloc((size_t)TGN_N * TGN_H * 4);
  float* AGG  = (float*)alloc((size_t)TGN_N * TGN_H * 4);
  float* deg  = (float*)alloc((size_t)TGN_N * 4);
  float* dinv = (float*)alloc((size_t)TGN_N * 4);

  const int MT = TGN_N / 16;  // 1250 row tiles
  dim3 wave(32);

  // One-time (per launch) weight transpose+convert
  tgn_prep_bt<<<(TGN_F * TGN_H + 255) / 256, 256, 0, stream>>>(W_in, Wt_in, TGN_F, TGN_H);
  tgn_prep_bt<<<(3 * TGN_H * TGN_H + 255) / 256, 256, 0, stream>>>(W_ih, Wt_ih, TGN_H, 3 * TGN_H);
  tgn_prep_bt<<<(3 * TGN_H * TGN_H + 255) / 256, 256, 0, stream>>>(W_hh, Wt_hh, TGN_H, 3 * TGN_H);
  for (int l = 0; l < TGN_L; ++l)
    tgn_prep_bt<<<(TGN_H * TGN_H + 255) / 256, 256, 0, stream>>>(
        conv_w + (size_t)l * TGN_H * TGN_H, Wt_conv + (size_t)l * TGN_H * TGN_H, TGN_H, TGN_H);

  // h0 = 0
  tgn_zero<<<(TGN_N * TGN_H + 255) / 256, 256, 0, stream>>>(Ha, TGN_N * TGN_H);

  float* Hcur = Ha;
  float* Halt = Hb;

  for (int t = 0; t < TGN_T; ++t) {
    const float* xt  = x_seq + (size_t)t * TGN_N * TGN_F;
    const int*   src = edge_index + (size_t)t * 2 * TGN_E;
    const int*   dst = src + TGN_E;
    const float* ewt = edge_weight + (size_t)t * TGN_E;

    // Xp = x_t @ W_in + b_in
    tgn_gemm<<<dim3(MT, TGN_H / 16), wave, 0, stream>>>(xt, Wt_in, b_in, Xp, TGN_F, TGN_H);

    // Fused GRU: Hcur -> Halt
    tgn_gru<<<dim3(MT, TGN_H / 16), wave, 0, stream>>>(Xp, Hcur, Wt_ih, Wt_hh, b_ih, b_hh, Halt);
    { float* tmp = Hcur; Hcur = Halt; Halt = tmp; }

    for (int l = 0; l < TGN_L; ++l) {
      // HW = Hcur @ conv_w[l]
      tgn_gemm<<<dim3(MT, TGN_H / 16), wave, 0, stream>>>(
          Hcur, Wt_conv + (size_t)l * TGN_H * TGN_H, nullptr, HW, TGN_H, TGN_H);
      // degree / dinv for this timestep's graph
      tgn_zero<<<(TGN_N + 255) / 256, 256, 0, stream>>>(deg, TGN_N);
      tgn_degree<<<(TGN_E + 255) / 256, 256, 0, stream>>>(dst, ewt, deg, TGN_E);
      tgn_dinv<<<(TGN_N + 255) / 256, 256, 0, stream>>>(deg, dinv, TGN_N);
      // edge aggregation
      tgn_zero<<<(TGN_N * TGN_H + 255) / 256, 256, 0, stream>>>(AGG, TGN_N * TGN_H);
      tgn_scatter<<<(unsigned)(((long long)TGN_E * TGN_H + 255) / 256), 256, 0, stream>>>(
          src, dst, ewt, dinv, HW, AGG, TGN_E);
      // Halt = relu(AGG + dinv^2*HW + conv_b[l]) + Hcur
      tgn_combine<<<(TGN_N * TGN_H + 255) / 256, 256, 0, stream>>>(
          AGG, HW, dinv, conv_b + (size_t)l * TGN_H, Hcur, Halt, TGN_N * TGN_H);
      { float* tmp = Hcur; Hcur = Halt; Halt = tmp; }
    }

    // out[t] = Hcur @ W_out + b_out
    tgn_out<<<(TGN_N * 32 + 255) / 256, 256, 0, stream>>>(Hcur, W_out, b_out, out + (size_t)t * TGN_N, TGN_N);
  }
}
